// ModulatedDeformConv2dLayerBlock_38482906972780
// MI455X (gfx1250) — matrix operand order repaired and checked by reading, not compiled, not run on hardware
//
#include <hip/hip_runtime.h>
#include <hip/hip_bf16.h>

#define BATCH 4
#define CIN   64
#define COUT  64
#define HI    128
#define WI    128
#define HO    126
#define WO    126
#define HW    (HO*WO)        // 15876
#define KKT   9              // taps
#define KDIM  (CIN*KKT)      // 576 (GEMM K)
#define TPITCH 584           // LDS row pitch in bf16 elems (584*2 = 1168 = 73*16)
#define NPIX  32             // pixels per workgroup (two 16-wide N tiles)
#define NPAIR (NPIX*KKT)     // 288 (pixel, tap) pairs per workgroup

typedef __attribute__((ext_vector_type(16))) __bf16       v16bf;
typedef __attribute__((ext_vector_type(8)))  float        v8f;
typedef __attribute__((ext_vector_type(4)))  unsigned int uint4v;

static __device__ __forceinline__ unsigned short f2bf(float f) {
    unsigned int u = __float_as_uint(f);
    u += 0x7fffu + ((u >> 16) & 1u);          // round-to-nearest-even
    return (unsigned short)(u >> 16);
}

// ---------------- Kernel 1: offset conv (27ch, 3x3, VALID) ----------------
__global__ void __launch_bounds__(256)
offset_conv_kernel(const float* __restrict__ x,
                   const float* __restrict__ w_off,
                   const float* __restrict__ b_off,
                   float* __restrict__ wsY,
                   float* __restrict__ wsX,
                   float* __restrict__ wsM) {
    // LDS copy of w_off, transposed to [(c*9+t)*27 + j] so inner j-loop is contiguous
    __shared__ float sw[27 * KDIM];
    const int tid = threadIdx.x;
    for (int idx = tid; idx < 27 * KDIM; idx += 256) {
        int j   = idx / KDIM;
        int rem = idx - j * KDIM;               // rem = c*9+t
        sw[rem * 27 + j] = w_off[idx];
    }
    __syncthreads();

    int p = blockIdx.x * 256 + tid;
    if (p >= BATCH * HW) return;
    int b  = p / HW;
    int pp = p - b * HW;
    int ho = pp / WO;
    int wo = pp - ho * WO;

    float acc[27];
    #pragma unroll
    for (int j = 0; j < 27; ++j) acc[j] = b_off[j];

    const float* xb = x + (size_t)b * CIN * HI * WI + (size_t)ho * WI + wo;
    for (int c = 0; c < CIN; ++c) {
        const float* xc = xb + (size_t)c * (HI * WI);
        float win[9];
        #pragma unroll
        for (int t = 0; t < 9; ++t)
            win[t] = xc[(t / 3) * WI + (t % 3)];
        const float* swc = &sw[c * 9 * 27];
        #pragma unroll
        for (int t = 0; t < 9; ++t) {
            const float xv = win[t];
            const float* swt = swc + t * 27;
            #pragma unroll
            for (int j = 0; j < 27; ++j)
                acc[j] = fmaf(xv, swt[j], acc[j]);
        }
    }

    // reference reshape interleaves: off_y[k]=ch[2k], off_x[k]=ch[2k+1], mask=ch[18+k]
    const size_t base = (size_t)b * KKT * HW + pp;
    #pragma unroll
    for (int k = 0; k < KKT; ++k) {
        wsY[base + (size_t)k * HW] = acc[2 * k];
        wsX[base + (size_t)k * HW] = acc[2 * k + 1];
        wsM[base + (size_t)k * HW] = 1.0f / (1.0f + __expf(-acc[18 + k]));
    }
}

// ---------------- Kernel 2: weight fp32 -> bf16 (A matrix, (64 x 576) row-major) ----
__global__ void wconv_kernel(const float* __restrict__ w,
                             unsigned short* __restrict__ wbf, int n) {
    int i = blockIdx.x * blockDim.x + threadIdx.x;
    if (i < n) wbf[i] = f2bf(w[i]);
}

// ---------------- Kernel 3: fused deform-sample + modulate + WMMA GEMM + bias/ReLU --
__global__ void __launch_bounds__(128)
mdconv_main_kernel(const float* __restrict__ x,
                   const unsigned short* __restrict__ wbf,
                   const float* __restrict__ bias,
                   const float* __restrict__ wsY,
                   const float* __restrict__ wsX,
                   const float* __restrict__ wsM,
                   float* __restrict__ out) {
    __shared__ int   s_y0[NPAIR], s_y1[NPAIR], s_x0[NPAIR], s_x1[NPAIR];
    __shared__ float s_w00[NPAIR], s_w01[NPAIR], s_w10[NPAIR], s_w11[NPAIR];
    __shared__ __align__(16) unsigned short s_smp[NPIX * TPITCH]; // 32 pixels x 576 bf16

    const int tid = threadIdx.x;
    const int tilesPerBatch = (HW + NPIX - 1) / NPIX;
    const int b = blockIdx.x / tilesPerBatch;
    const int pixBase = (blockIdx.x % tilesPerBatch) * NPIX;

    // ---- Stage 0: bilinear params per (pixel, tap), mask folded into weights ----
    for (int idx = tid; idx < NPAIR; idx += 128) {
        int pix = idx / KKT;
        int k   = idx - pix * KKT;
        int p   = pixBase + pix;
        int y0 = 0, y1 = 0, x0 = 0, x1 = 0;
        float w00 = 0.f, w01 = 0.f, w10 = 0.f, w11 = 0.f;
        if (p < HW) {
            int ho = p / WO;
            int wo = p - ho * WO;
            size_t oidx = ((size_t)(b * KKT + k)) * HW + p;
            float offY = wsY[oidx], offX = wsX[oidx], mk = wsM[oidx];
            float ys = (float)(ho + k / 3) + offY;
            float xs = (float)(wo + k % 3) + offX;
            float y0f = floorf(ys), x0f = floorf(xs);
            float wy = ys - y0f,   wx = xs - x0f;
            int iy0 = (int)y0f, ix0 = (int)x0f;
            int iy1 = iy0 + 1,  ix1 = ix0 + 1;
            bool vy0 = (iy0 >= 0) & (iy0 < HI);
            bool vy1 = (iy1 >= 0) & (iy1 < HI);
            bool vx0 = (ix0 >= 0) & (ix0 < WI);
            bool vx1 = (ix1 >= 0) & (ix1 < WI);
            w00 = (vy0 & vx0) ? (1.f - wy) * (1.f - wx) * mk : 0.f;
            w01 = (vy0 & vx1) ? (1.f - wy) * wx         * mk : 0.f;
            w10 = (vy1 & vx0) ? wy         * (1.f - wx) * mk : 0.f;
            w11 = (vy1 & vx1) ? wy         * wx         * mk : 0.f;
            y0 = iy0 < 0 ? 0 : (iy0 > HI - 1 ? HI - 1 : iy0);
            y1 = iy1 < 0 ? 0 : (iy1 > HI - 1 ? HI - 1 : iy1);
            x0 = ix0 < 0 ? 0 : (ix0 > WI - 1 ? WI - 1 : ix0);
            x1 = ix1 < 0 ? 0 : (ix1 > WI - 1 ? WI - 1 : ix1);
        }
        s_y0[idx] = y0; s_y1[idx] = y1; s_x0[idx] = x0; s_x1[idx] = x1;
        s_w00[idx] = w00; s_w01[idx] = w01; s_w10[idx] = w10; s_w11[idx] = w11;
    }
    __syncthreads();

    // ---- Stage 1: gather + modulate sampled tile into LDS as bf16 ----
    // B matrix layout: row = pixel (N), col = c*9+k (K), pitch TPITCH
    for (int c = 0; c < CIN; ++c) {
        const float* xc = x + ((size_t)(b * CIN + c)) * (HI * WI);
        for (int pair = tid; pair < NPAIR; pair += 128) {
            int pix = pair / KKT;
            int k   = pair - pix * KKT;
            int y0 = s_y0[pair], y1 = s_y1[pair], x0 = s_x0[pair], x1 = s_x1[pair];
            float v = s_w00[pair] * xc[y0 * WI + x0]
                    + s_w01[pair] * xc[y0 * WI + x1]
                    + s_w10[pair] * xc[y1 * WI + x0]
                    + s_w11[pair] * xc[y1 * WI + x1];
            s_smp[pix * TPITCH + (c * KKT + k)] = f2bf(v);
        }
    }
    __syncthreads();

    // ---- Stage 2: each wave = 16 Cout rows x 32 pixels (two 16x16 tiles / K step) --
    // A-fragment (weight) is loaded once per K step and feeds BOTH N tiles.
    const int lane = tid & 31;
    const int wave = tid >> 5;
    const int coutBase = wave * 16;
    const int mA = lane & 15;            // A row (M)
    const int kA = (lane >> 4) * 16;     // A K half-select (contiguous, matches B)
    const int nB = lane & 15;            // B col (N) within tile
    const int kB = (lane >> 4) * 16;     // B K half-select

    union Frag { uint4v q[2]; v16bf v; };
    v8f acc0 = {};
    v8f acc1 = {};

    const unsigned short* wrow  = wbf + (size_t)(coutBase + mA) * KDIM;
    const unsigned short* srow0 = &s_smp[nB * TPITCH];
    const unsigned short* srow1 = &s_smp[(16 + nB) * TPITCH];

    for (int kk = 0; kk < KDIM; kk += 32) {
        Frag a, b0, b1;
        a.q[0]  = *(const uint4v*)(wrow + kk + kA);        // elems 0..7  -> K = kk+kA+0..7
        a.q[1]  = *(const uint4v*)(wrow + kk + kA + 8);    // elems 8..15 -> K = kk+kA+8..15
        b0.q[0] = *(const uint4v*)(srow0 + kk + kB);
        b0.q[1] = *(const uint4v*)(srow0 + kk + kB + 8);
        b1.q[0] = *(const uint4v*)(srow1 + kk + kB);
        b1.q[1] = *(const uint4v*)(srow1 + kk + kB + 8);
        acc0 = __builtin_amdgcn_wmma_f32_16x16x32_bf16(false, a.v, false, b0.v,
                                                       (short)0, acc0, false, false);
        acc1 = __builtin_amdgcn_wmma_f32_16x16x32_bf16(false, a.v, false, b1.v,
                                                       (short)0, acc1, false, false);
    }

    // ---- Epilogue: bias + ReLU, C layout: n = lane&15, m = (lane>>4)*8 + r ----
    const int mBase = (lane >> 4) * 8;
    const int p0 = pixBase + nB;
    const int p1 = pixBase + 16 + nB;
    #pragma unroll
    for (int r = 0; r < 8; ++r) {
        const int o = coutBase + mBase + r;
        const float bi = bias[o];
        const size_t rowOff = ((size_t)(b * COUT + o)) * HW;
        if (p0 < HW) {
            float v = acc0[r] + bi;
            out[rowOff + p0] = v > 0.f ? v : 0.f;
        }
        if (p1 < HW) {
            float v = acc1[r] + bi;
            out[rowOff + p1] = v > 0.f ? v : 0.f;
        }
    }
}

// ---------------- Host launch ----------------
extern "C" void kernel_launch(void* const* d_in, const int* in_sizes, int n_in,
                              void* d_out, int out_size, void* d_ws, size_t ws_size,
                              hipStream_t stream) {
    const float* x      = (const float*)d_in[0];
    const float* w_off  = (const float*)d_in[1];
    const float* b_off  = (const float*)d_in[2];
    const float* weight = (const float*)d_in[3];
    const float* bias   = (const float*)d_in[4];
    float* out = (float*)d_out;

    const size_t S = (size_t)BATCH * KKT * HW;       // per-map elements
    float* wsY = (float*)d_ws;
    float* wsX = wsY + S;
    float* wsM = wsX + S;
    unsigned short* wbf = (unsigned short*)(wsM + S); // 3*S*4 bytes = 16B aligned

    const int np = BATCH * HW;
    offset_conv_kernel<<<(np + 255) / 256, 256, 0, stream>>>(x, w_off, b_off, wsY, wsX, wsM);

    const int nw = COUT * KDIM;
    wconv_kernel<<<(nw + 255) / 256, 256, 0, stream>>>(weight, wbf, nw);

    const int tilesPerBatch = (HW + NPIX - 1) / NPIX; // 497 (ragged last tile handled)
    mdconv_main_kernel<<<BATCH * tilesPerBatch, 128, 0, stream>>>(
        x, wbf, bias, wsY, wsX, wsM, out);
}
